// Multihead_Self_Attention_8881992368179
// MI455X (gfx1250) — matrix operand order repaired and checked by reading, not compiled, not run on hardware
//
#include <hip/hip_runtime.h>
#include <hip/hip_bf16.h>
#include <math.h>

// ---------------------------------------------------------------------------
// MHA with RoPE on gfx1250 (MI455X).  Compute-bound (~100 GFLOP vs ~100 MB):
// all GEMMs go through v_wmma_f32_16x16x32_bf16 with fp32 accumulation.
// Attention stages K/V chunks into LDS once per block (4 waves share them)
// via GLOBAL_LOAD_ASYNC_TO_LDS_B128 (ASYNCcnt), killing 4x redundant VMEM.
// ---------------------------------------------------------------------------

typedef __attribute__((ext_vector_type(16))) __bf16 v16bf;
typedef __attribute__((ext_vector_type(8)))  float  v8f;
typedef __attribute__((ext_vector_type(4)))  int    v4i;

#define B_   2
#define N_   2048
#define DM_  1024
#define H_   16
#define D_   64

#if __has_builtin(__builtin_amdgcn_global_load_async_to_lds_b128)
#define ASYNC_OK 1
#endif

__device__ __forceinline__ void async_cp16(const __bf16* g, __bf16* l) {
#ifdef ASYNC_OK
  // param0: AS1 (global) v4i*, param1: AS3 (LDS) v4i*, then imm offset, cpol
  __builtin_amdgcn_global_load_async_to_lds_b128(
      (__attribute__((address_space(1))) v4i*)(void*)g,
      (__attribute__((address_space(3))) v4i*)(void*)l, 0, 0);
#else
  *(uint4*)l = *(const uint4*)g;
#endif
}

__device__ __forceinline__ void wait_async0() {
#ifdef ASYNC_OK
#if __has_builtin(__builtin_amdgcn_s_wait_asynccnt)
  __builtin_amdgcn_s_wait_asynccnt(0);
#else
  asm volatile("s_wait_asynccnt 0" ::: "memory");
#endif
#endif
}

__device__ __forceinline__ v8f wmma_bf16(v16bf a, v16bf b, v8f c) {
  return __builtin_amdgcn_wmma_f32_16x16x32_bf16(false, a, false, b,
                                                 (short)0, c, false, false);
}

// A operand, 16x32 bf16 (M x K).  `row` = this lane's M-row, 32 contiguous K.
// ISA layout: lanes 0-15 hold K={0..7,16..23}; lanes 16-31 hold K={8..15,24..31}.
__device__ __forceinline__ v16bf load_a(const __bf16* __restrict__ row) {
  const int hi = ((threadIdx.x & 31) >> 4) * 8;
  v16bf a;
#pragma unroll
  for (int i = 0; i < 8; ++i) a[i] = row[hi + i];
#pragma unroll
  for (int i = 0; i < 8; ++i) a[8 + i] = row[16 + hi + i];
  return a;
}

// B operand, 32x16 bf16 (K x N).  `col` = this lane's N-column as a pointer to
// 32 contiguous K values.  Lanes 0-15 hold K={0..7,16..23}; lanes 16-31 hold K={8..15,24..31}.
__device__ __forceinline__ v16bf load_b(const __bf16* __restrict__ col) {
  const __bf16* p = col + (((threadIdx.x & 31) >> 4) ? 8 : 0);
  v16bf b;
#pragma unroll
  for (int i = 0; i < 8; ++i) { b[i] = p[i]; b[8 + i] = p[16 + i]; }
  return b;
}

__device__ __forceinline__ float rowmax16(float x) {
  x = fmaxf(x, __shfl_xor(x, 1));
  x = fmaxf(x, __shfl_xor(x, 2));
  x = fmaxf(x, __shfl_xor(x, 4));
  x = fmaxf(x, __shfl_xor(x, 8));
  return x;
}
__device__ __forceinline__ float rowsum16(float x) {
  x += __shfl_xor(x, 1);
  x += __shfl_xor(x, 2);
  x += __shfl_xor(x, 4);
  x += __shfl_xor(x, 8);
  return x;
}

// --------------------------- fp32 -> bf16 convert ---------------------------
__global__ void cvt_bf16_kernel(const float* __restrict__ src,
                                __bf16* __restrict__ dst, int n) {
  int i = blockIdx.x * blockDim.x + threadIdx.x;
  if (i < n) dst[i] = (__bf16)src[i];
}

// --------------------- fused QKV projection + RoPE --------------------------
// grid (256, 3): x = M-tile (B*N/16), y = matrix (0=Q,1=K,2=V).
// 512 threads = 16 waves; wave w computes the 16x64 strip of head h=w.
__global__ __launch_bounds__(512)
void qkv_rope_kernel(const __bf16* __restrict__ xb,
                     const __bf16* __restrict__ wb,
                     const int*    __restrict__ tp,
                     __bf16* __restrict__ Qb,   // [B,H,N,D]
                     __bf16* __restrict__ Kb,   // [B,H,N,D]
                     __bf16* __restrict__ Vt) { // [B,H,D,N]
  const int lane  = threadIdx.x & 31;
  const int h     = threadIdx.x >> 5;          // wave = head = 64-col strip
  const int mbase = blockIdx.x * 16;
  const int mat   = blockIdx.y;
  const int hi    = lane >> 4;
  const int ln    = lane & 15;

  const __bf16* __restrict__ W    = wb + (size_t)mat * (DM_ * DM_);
  const __bf16* __restrict__ arow = xb + (size_t)(mbase + ln) * DM_;

  v8f acc[4] = {};                             // 16x64 fp32 accumulators
  for (int k = 0; k < DM_; k += 32) {
    v16bf a = load_a(arow + k);
#pragma unroll
    for (int c = 0; c < 4; ++c) {
      // y = x * W^T  =>  Bmat[k][j] = W[j][k], contiguous along k
      v16bf bop = load_b(W + (size_t)(h * 64 + c * 16 + ln) * DM_ + k);
      acc[c] = wmma_bf16(a, bop, acc[c]);
    }
  }

  if (mat == 2) {                              // V: store transposed [b,h,d,n]
#pragma unroll
    for (int c = 0; c < 4; ++c)
#pragma unroll
      for (int r = 0; r < 8; ++r) {
        int m = mbase + r + 8 * hi;
        int bb = m >> 11, n = m & (N_ - 1);
        int d = c * 16 + ln;
        Vt[(((size_t)(bb * H_ + h)) * D_ + d) * N_ + n] = (__bf16)acc[c][r];
      }
  } else {                                     // Q/K: RoPE then store [b,h,n,d]
    __bf16* __restrict__ dst = (mat == 0) ? Qb : Kb;
#pragma unroll
    for (int r = 0; r < 8; ++r) {
      int m = mbase + r + 8 * hi;
      int bb = m >> 11, n = m & (N_ - 1);
      float pos = (float)tp[n];
#pragma unroll
      for (int c = 0; c < 4; ++c) {
        int   d  = c * 16 + ln;
        // inv_freq = theta^(-2t/D), t = d/2 ; ln(10000)/64 = 0.143911568f
        float inv_freq = __expf(-(float)(d & ~1) * 0.143911568f);
        float sn, cs;
        __sincosf(pos * inv_freq, &sn, &cs);
        float x   = acc[c][r];
        float prt = __shfl_xor(x, 1);          // rotary partner = adjacent lane
        float res = (lane & 1) ? (prt * sn + x * cs)   // odd:  x1*s + x2*c
                               : (x * cs - prt * sn);  // even: x1*c - x2*s
        dst[(((size_t)(bb * H_ + h)) * N_ + n) * D_ + d] = (__bf16)res;
      }
    }
  }
}

// --------------------- causal flash attention core --------------------------
// grid 1024 x 128 threads (4 waves).  The 4 waves handle 4 consecutive query
// tiles of the SAME (b,h): each 32-key K/V chunk is async-staged into LDS once
// per block and shared by all 4 waves.
__global__ __launch_bounds__(128)
void attn_kernel(const __bf16* __restrict__ Qb,
                 const __bf16* __restrict__ Kb,
                 const __bf16* __restrict__ Vt,
                 __bf16* __restrict__ Ob) {    // [B*N, DM] bf16
  __shared__ __align__(16) __bf16 kbuf[32 * 64];    // K chunk  [j-jb][d]
  __shared__ __align__(16) __bf16 vbuf[64 * 32];    // Vt chunk [d][j-jb]
  __shared__ __align__(16) __bf16 pt[4][16 * 32];   // per-wave P transpose pad

  const int tid  = threadIdx.x;
  const int lane = tid & 31;
  const int w    = tid >> 5;
  const int gw   = blockIdx.x * 4 + w;         // 0..4095
  const int qt   = gw & 127;                   // query tile
  const int h    = (gw >> 7) & 15;
  const int b    = gw >> 11;
  const int hi   = lane >> 4, ln = lane & 15;
  const int ibase     = qt * 16;               // wave-uniform
  const int ibase_max = (((blockIdx.x * 4) & 127) + 3) * 16;  // block-uniform
  const size_t bh = (size_t)(b * H_ + h);
  const __bf16* __restrict__ kbase = Kb + bh * N_ * D_;
  const __bf16* __restrict__ vbase = Vt + bh * D_ * N_;

  const __bf16* __restrict__ qrow = Qb + (bh * N_ + ibase + ln) * D_;
  v16bf a0 = load_a(qrow);                     // d = 0..31
  v16bf a1 = load_a(qrow + 32);                // d = 32..63

  v8f   o[4] = {};
  float m_r[8], l_r[8];
#pragma unroll
  for (int r = 0; r < 8; ++r) { m_r[r] = -1e30f; l_r[r] = 0.f; }

  for (int jb = 0; jb <= ibase_max; jb += 32) {
    // ---- cooperative async stage: K 32x64 + V 64x32 (4 KB each) ----
#pragma unroll
    for (int t = 0; t < 2; ++t) {
      int c = tid + t * 128;                   // 0..255 16-byte chunks each
      int kr = c >> 3, kp = c & 7;             // K: row j-jb, 16B part of d
      int krow = jb + kr; krow = krow < N_ - 1 ? krow : N_ - 1;  // clamp tail
      async_cp16(kbase + (size_t)krow * D_ + kp * 8, &kbuf[kr * 64 + kp * 8]);
      int vd = c >> 2, vp = c & 3;             // V: d row, 16B part of j
      int vcol = jb + vp * 8; vcol = vcol < N_ - 8 ? vcol : N_ - 8;
      async_cp16(vbase + (size_t)vd * N_ + vcol, &vbuf[vd * 32 + vp * 8]);
    }
    wait_async0();
    __syncthreads();

    if (jb <= ibase) {                         // wave-uniform: EXEC stays full
      // ---- scores S = Q K^T (contraction over d=64, K from LDS) ----
      v8f s[2];
#pragma unroll
      for (int jt = 0; jt < 2; ++jt) {
        const __bf16* krow = &kbuf[(jt * 16 + ln) * 64];
        v8f c = {};
        c = wmma_bf16(a0, load_b(krow), c);
        c = wmma_bf16(a1, load_b(krow + 32), c);
        s[jt] = c;
      }
      // ---- causal mask + scale (1/sqrt(64)) ----
#pragma unroll
      for (int jt = 0; jt < 2; ++jt) {
        int j = jb + jt * 16 + ln;
#pragma unroll
        for (int r = 0; r < 8; ++r) {
          int i = ibase + r + 8 * hi;
          s[jt][r] = (j <= i) ? s[jt][r] * 0.125f : -1e30f;
        }
      }
      // ---- online softmax (row reductions across 16-lane halves) ----
#pragma unroll
      for (int r = 0; r < 8; ++r) {
        float rm   = rowmax16(fmaxf(s[0][r], s[1][r]));
        float mn   = fmaxf(m_r[r], rm);
        float corr = __expf(m_r[r] - mn);
        float p0   = __expf(s[0][r] - mn);
        float p1   = __expf(s[1][r] - mn);
        s[0][r] = p0; s[1][r] = p1;
        l_r[r] = l_r[r] * corr + rowsum16(p0 + p1);
        m_r[r] = mn;
#pragma unroll
        for (int c = 0; c < 4; ++c) o[c][r] *= corr;
      }
      // ---- C-layout -> A-layout transpose of P through LDS (in-order DS) ----
#pragma unroll
      for (int jt = 0; jt < 2; ++jt)
#pragma unroll
        for (int r = 0; r < 8; ++r)
          pt[w][(r + 8 * hi) * 32 + jt * 16 + ln] = (__bf16)s[jt][r];
      asm volatile("s_wait_dscnt 0" ::: "memory");
      v16bf pa = load_a(&pt[w][ln * 32]);
      // ---- O += P @ V  (V from LDS, contiguous along j) ----
#pragma unroll
      for (int c = 0; c < 4; ++c)
        o[c] = wmma_bf16(pa, load_b(&vbuf[(c * 16 + ln) * 32]), o[c]);
    }
    __syncthreads();                           // protect buffers for next chunk
  }

  // ---- normalize and store bf16 in [B*N, H*D] for the output projection ----
#pragma unroll
  for (int c = 0; c < 4; ++c)
#pragma unroll
    for (int r = 0; r < 8; ++r) {
      int n = ibase + r + 8 * hi;
      Ob[((size_t)b * N_ + n) * DM_ + h * 64 + c * 16 + ln] =
          (__bf16)(o[c][r] / l_r[r]);
    }
}

// --------------------------- output projection ------------------------------
__global__ __launch_bounds__(512)
void oproj_kernel(const __bf16* __restrict__ Ob,
                  const __bf16* __restrict__ Wo,
                  float* __restrict__ out) {
  const int lane  = threadIdx.x & 31;
  const int strip = threadIdx.x >> 5;          // 16 waves -> 16 x 64-col strips
  const int mbase = blockIdx.x * 16;
  const int hi    = lane >> 4, ln = lane & 15;

  const __bf16* __restrict__ arow = Ob + (size_t)(mbase + ln) * DM_;
  v8f acc[4] = {};
  for (int k = 0; k < DM_; k += 32) {
    v16bf a = load_a(arow + k);
#pragma unroll
    for (int c = 0; c < 4; ++c) {
      v16bf bop = load_b(Wo + (size_t)(strip * 64 + c * 16 + ln) * DM_ + k);
      acc[c] = wmma_bf16(a, bop, acc[c]);
    }
  }
#pragma unroll
  for (int c = 0; c < 4; ++c)
#pragma unroll
    for (int r = 0; r < 8; ++r)
      out[(size_t)(mbase + r + 8 * hi) * DM_ + strip * 64 + c * 16 + ln] =
          acc[c][r];
}

// ---------------------------------------------------------------------------
extern "C" void kernel_launch(void* const* d_in, const int* in_sizes, int n_in,
                              void* d_out, int out_size, void* d_ws,
                              size_t ws_size, hipStream_t stream) {
  const float* x  = (const float*)d_in[0];
  const float* Wq = (const float*)d_in[1];
  const float* Wk = (const float*)d_in[2];
  const float* Wv = (const float*)d_in[3];
  const float* Wo = (const float*)d_in[4];
  const int*   tp = (const int*)d_in[5];
  float* out = (float*)d_out;

  const size_t MN  = (size_t)B_ * N_;          // 4096 rows
  const size_t XE  = MN * DM_;                 // 4 Mi elements
  const size_t WE  = (size_t)DM_ * DM_;        // 1 Mi elements per weight
  const size_t QKV = (size_t)B_ * H_ * N_ * D_;

  __bf16* xb = (__bf16*)d_ws;                  // x in bf16
  __bf16* wb = xb + XE;                        // Wq|Wk|Wv|Wo in bf16
  __bf16* Qb = wb + 4 * WE;                    // [B,H,N,D]
  __bf16* Kb = Qb + QKV;                       // [B,H,N,D]
  __bf16* Vt = Kb + QKV;                       // [B,H,D,N]
  __bf16* Ob = Vt + QKV;                       // [B*N, DM]

  cvt_bf16_kernel<<<(int)((XE + 255) / 256), 256, 0, stream>>>(x,  xb, (int)XE);
  cvt_bf16_kernel<<<(int)((WE + 255) / 256), 256, 0, stream>>>(Wq, wb + 0 * WE, (int)WE);
  cvt_bf16_kernel<<<(int)((WE + 255) / 256), 256, 0, stream>>>(Wk, wb + 1 * WE, (int)WE);
  cvt_bf16_kernel<<<(int)((WE + 255) / 256), 256, 0, stream>>>(Wv, wb + 2 * WE, (int)WE);
  cvt_bf16_kernel<<<(int)((WE + 255) / 256), 256, 0, stream>>>(Wo, wb + 3 * WE, (int)WE);

  qkv_rope_kernel<<<dim3(MN / 16, 3), 512, 0, stream>>>(xb, wb, tp, Qb, Kb, Vt);
  attn_kernel<<<(B_ * H_ * (N_ / 16)) / 4, 128, 0, stream>>>(Qb, Kb, Vt, Ob);
  oproj_kernel<<<MN / 16, 512, 0, stream>>>(Ob, wb + 3 * WE, out);
}
